// MultiHeadSelfAttention_74285754351784
// MI455X (gfx1250) — compile-verified, matrix operand order checked by reading, not run on hardware
//
#include <hip/hip_runtime.h>
#include <hip/hip_bf16.h>
#include <math.h>

// ---------- problem constants ----------
#define BB 4
#define SS 2048
#define DD 1024
#define HH 16
#define HD 64
#define M_TOT (BB * SS)        // 8192 rows
#define N_QKV (3 * DD)         // 3072

typedef __attribute__((ext_vector_type(16))) _Float16 v16h;
typedef __attribute__((ext_vector_type(8)))  float    v8f;
typedef __attribute__((ext_vector_type(4)))  unsigned int v4u;

// ---------- WMMA fragment helpers ----------
// 16x32 f16 A/B fragment (8 VGPRs), row/col contiguous along K.
// Lanes 0-15: K=0..7 (v0-3), K=16..23 (v4-7); lanes 16-31: K=8..15, K=24..31.
__device__ __forceinline__ v16h load_frag(const _Float16* p, int lh) {
    union { v16h h; v4u u[2]; } f;
    f.u[0] = *reinterpret_cast<const v4u*>(p + lh);
    f.u[1] = *reinterpret_cast<const v4u*>(p + lh + 16);
    return f.h;
}

// Plain (non-volatile) loads: pointers rooted at __shared__ arrays are
// address-space-inferred to LDS and lower to ds_load_b128.  The tiles are
// written by the inline-asm async DMA; their address escapes via ptrtoint
// into the asm (which has a "memory" clobber), so the loads are not folded.
__device__ __forceinline__ v16h load_frag_lds(const _Float16* p, int lh) {
    union { v16h h; v4u u[2]; } f;
    f.u[0] = *reinterpret_cast<const v4u*>(p + lh);
    f.u[1] = *reinterpret_cast<const v4u*>(p + lh + 16);
    return f.h;
}

__device__ __forceinline__ v8f wmma_f16(v16h a, v16h b, v8f c) {
    return __builtin_amdgcn_wmma_f32_16x16x32_f16(false, a, false, b,
                                                  (short)0, c, false, false);
}

// ---------- CDNA5 async global->LDS copy (ASYNCcnt path) ----------
__device__ __forceinline__ void async_copy_b128(unsigned lds_addr, const void* g) {
    unsigned long long ga = (unsigned long long)(uintptr_t)g;
    asm volatile("global_load_async_to_lds_b128 %0, %1, off"
                 :: "v"(lds_addr), "v"(ga) : "memory");
}
__device__ __forceinline__ void wait_async0() {
    asm volatile("s_wait_asynccnt 0" ::: "memory");
}

// Fill a 4096-byte LDS tile (rows of (1<<wlog2) bytes) from global rows of
// stride g_stride bytes.  128 threads x 2 x 16B = 4096B.
__device__ __forceinline__ void async_fill_tile(unsigned lds_base, const char* gbase,
                                                int wlog2, size_t g_stride, int tid) {
    #pragma unroll
    for (int s = 0; s < 2; ++s) {
        int b   = tid * 16 + s * 2048;
        int row = b >> wlog2;
        int ko  = b & ((1 << wlog2) - 1);
        async_copy_b128(lds_base + b, gbase + (size_t)row * g_stride + ko);
    }
}

// ---------- conversion kernels ----------
__global__ void cvt_f32_to_f16(const float* __restrict__ in,
                               _Float16* __restrict__ out, int n) {
    int i = blockIdx.x * blockDim.x + threadIdx.x;
    if (i < n) out[i] = (_Float16)in[i];
}

// out[n*rows + k] = in[k*cols + n]  (transpose + convert)
__global__ void transpose_cvt(const float* __restrict__ in,
                              _Float16* __restrict__ out, int rows, int cols) {
    int i = blockIdx.x * blockDim.x + threadIdx.x;
    if (i < rows * cols) {
        int n = i / rows;
        int k = i - n * rows;
        out[i] = (_Float16)in[k * cols + n];
    }
}

// ---------- GEMM1: qkv = src @ qkv_w + qkv_b, scattered into Q,K,V^T ----------
// Block tile 128(M) x 64(N); 4 waves, each wave 32x64 (8 WMMA / k-chunk).
// B tile (64 cols x 32 K halves = 4KB) staged async into LDS, double buffered.
__global__ void __launch_bounds__(128)
qkv_gemm_kernel(const _Float16* __restrict__ A,    // [8192,1024]
                const _Float16* __restrict__ BT,   // [3072,1024] = W^T
                const float*    __restrict__ bias, // [3072]
                _Float16* __restrict__ qd,   // [B*H, S, HD]
                _Float16* __restrict__ kd,   // [B*H, S, HD]
                _Float16* __restrict__ vtd)  // [B*H, HD, S]
{
    __shared__ __align__(16) _Float16 ldsB[2][64][32];   // 8 KB

    const int tid  = threadIdx.x;
    const int lane = tid & 31;
    const int wave = tid >> 5;
    const int n0   = blockIdx.x * 64;
    const int m0   = blockIdx.y * 128 + wave * 32;
    const int r    = lane & 15;
    const int lh   = (lane & 16) ? 8 : 0;
    const int hs   = (lane >> 4) & 1;

    const unsigned ldsb = (unsigned)(uintptr_t)&ldsB[0][0][0];
    const _Float16* arow0 = A + (size_t)(m0 + r) * DD;
    const _Float16* arow1 = A + (size_t)(m0 + 16 + r) * DD;

    async_fill_tile(ldsb, (const char*)(BT + (size_t)n0 * DD), 6, (size_t)DD * 2, tid);
    wait_async0();
    __syncthreads();

    v16h a0c = load_frag(arow0, lh);
    v16h a1c = load_frag(arow1, lh);

    v8f acc0[4] = {}, acc1[4] = {};

    for (int kc = 0; kc < DD; kc += 32) {
        const int  buf  = (kc >> 5) & 1;
        const bool more = (kc + 32) < DD;
        if (more)
            async_fill_tile(ldsb + (buf ^ 1) * 4096,
                            (const char*)(BT + (size_t)n0 * DD + kc + 32),
                            6, (size_t)DD * 2, tid);
        v16h a0n = a0c, a1n = a1c;
        if (more) {
            a0n = load_frag(arow0 + kc + 32, lh);
            a1n = load_frag(arow1 + kc + 32, lh);
        }
        #pragma unroll
        for (int nt = 0; nt < 4; ++nt) {
            v16h bf = load_frag_lds(&ldsB[buf][nt * 16 + r][0], lh);
            acc0[nt] = wmma_f16(a0c, bf, acc0[nt]);
            acc1[nt] = wmma_f16(a1c, bf, acc1[nt]);
        }
        a0c = a0n; a1c = a1n;
        wait_async0();
        __syncthreads();
    }

    #pragma unroll
    for (int nt = 0; nt < 4; ++nt) {
        const int col  = n0 + nt * 16 + r;
        const float bc = bias[col];
        const int h    = col / (3 * HD);
        const int rr   = col - h * (3 * HD);
        const int sel  = rr >> 6;              // 0=q 1=k 2=v
        const int hd   = rr & 63;
        #pragma unroll
        for (int mt = 0; mt < 2; ++mt) {
            const v8f* accp = mt ? acc1 : acc0;
            #pragma unroll
            for (int g = 0; g < 8; ++g) {
                const int row = m0 + mt * 16 + g + hs * 8;
                const int b   = row >> 11;
                const int s   = row & (SS - 1);
                const _Float16 hv = (_Float16)(accp[nt][g] + bc);
                const int bh = b * HH + h;
                if (sel == 0)      qd [((size_t)bh * SS + s) * HD + hd] = hv;
                else if (sel == 1) kd [((size_t)bh * SS + s) * HD + hd] = hv;
                else               vtd[((size_t)bh * HD + hd) * SS + s] = hv;
            }
        }
    }
}

// ---------- attention: flash-style, K/V tiles async-staged to LDS ----------
__global__ void __launch_bounds__(128)
attn_kernel(const _Float16* __restrict__ q,    // [BH, S, HD]
            const _Float16* __restrict__ k,    // [BH, S, HD]
            const _Float16* __restrict__ vt,   // [BH, HD, S]
            _Float16* __restrict__ attn)       // [B, S, H*HD]
{
    __shared__ __align__(16) _Float16 ldsK[2][32][64];   // 8 KB
    __shared__ __align__(16) _Float16 ldsV[2][64][32];   // 8 KB
    __shared__ __align__(16) _Float16 plds[4][16][32];   // 4 KB (per-wave P)

    const int tid  = threadIdx.x;
    const int lane = tid & 31;
    const int wave = tid >> 5;
    const int bh   = blockIdx.y;
    const int b    = bh >> 4;
    const int h    = bh & 15;
    const int q0   = blockIdx.x * 64 + wave * 16;
    const int r    = lane & 15;
    const int lh   = (lane & 16) ? 8 : 0;
    const int hs   = (lane >> 4) & 1;

    const _Float16* qb = q  + (size_t)bh * SS * HD;
    const _Float16* kb = k  + (size_t)bh * SS * HD;
    const _Float16* vb = vt + (size_t)bh * HD * SS;
    const unsigned ldsk = (unsigned)(uintptr_t)&ldsK[0][0][0];
    const unsigned ldsv = (unsigned)(uintptr_t)&ldsV[0][0][0];

    async_fill_tile(ldsk, (const char*)kb, 7, 128, tid);
    async_fill_tile(ldsv, (const char*)vb, 6, (size_t)SS * 2, tid);
    wait_async0();
    __syncthreads();

    const v16h qf0 = load_frag(qb + (size_t)(q0 + r) * HD +  0, lh);
    const v16h qf1 = load_frag(qb + (size_t)(q0 + r) * HD + 32, lh);

    v8f acc[4] = {};
    float m[8], l[8];
    #pragma unroll
    for (int g = 0; g < 8; ++g) { m[g] = -1e30f; l[g] = 0.0f; }

    const float scale = 0.125f;  // 1/sqrt(HD)

    for (int kv = 0; kv < SS; kv += 32) {
        const int  buf  = (kv >> 5) & 1;
        const bool more = (kv + 32) < SS;
        if (more) {
            async_fill_tile(ldsk + (buf ^ 1) * 4096,
                            (const char*)(kb + (size_t)(kv + 32) * HD), 7, 128, tid);
            async_fill_tile(ldsv + (buf ^ 1) * 4096,
                            (const char*)(vb + kv + 32), 6, (size_t)SS * 2, tid);
        }
        // scores 16q x 32k (two 16x16 C fragments) from LDS K tile
        v8f s0 = {}, s1 = {};
        s0 = wmma_f16(qf0, load_frag_lds(&ldsK[buf][r][0],       lh), s0);
        s0 = wmma_f16(qf1, load_frag_lds(&ldsK[buf][r][32],      lh), s0);
        s1 = wmma_f16(qf0, load_frag_lds(&ldsK[buf][16 + r][0],  lh), s1);
        s1 = wmma_f16(qf1, load_frag_lds(&ldsK[buf][16 + r][32], lh), s1);

        // prefetch V fragments (latency hides under the softmax VALU block)
        v16h vf0 = load_frag_lds(&ldsV[buf][ 0 + r][0], lh);
        v16h vf1 = load_frag_lds(&ldsV[buf][16 + r][0], lh);
        v16h vf2 = load_frag_lds(&ldsV[buf][32 + r][0], lh);
        v16h vf3 = load_frag_lds(&ldsV[buf][48 + r][0], lh);

        // online softmax: row stats reduce over the 16 lanes of each half
        #pragma unroll
        for (int g = 0; g < 8; ++g) {
            float a0 = s0[g] * scale;
            float a1 = s1[g] * scale;
            float mx = fmaxf(a0, a1);
            mx = fmaxf(mx, __shfl_xor(mx, 1));
            mx = fmaxf(mx, __shfl_xor(mx, 2));
            mx = fmaxf(mx, __shfl_xor(mx, 4));
            mx = fmaxf(mx, __shfl_xor(mx, 8));
            const float nm = fmaxf(m[g], mx);
            const float p0 = __expf(a0 - nm);
            const float p1 = __expf(a1 - nm);
            float rs = p0 + p1;
            rs += __shfl_xor(rs, 1);
            rs += __shfl_xor(rs, 2);
            rs += __shfl_xor(rs, 4);
            rs += __shfl_xor(rs, 8);
            const float alpha = __expf(m[g] - nm);
            l[g] = l[g] * alpha + rs;
            m[g] = nm;
            const int row = g + hs * 8;
            plds[wave][row][r]      = (_Float16)p0;
            plds[wave][row][16 + r] = (_Float16)p1;
            #pragma unroll
            for (int t = 0; t < 4; ++t) acc[t][g] *= alpha;
        }
        // re-read P in A-operand layout (wave-private; DS ops are in-order)
        const v16h pf = load_frag(&plds[wave][r][0], lh);

        acc[0] = wmma_f16(pf, vf0, acc[0]);
        acc[1] = wmma_f16(pf, vf1, acc[1]);
        acc[2] = wmma_f16(pf, vf2, acc[2]);
        acc[3] = wmma_f16(pf, vf3, acc[3]);

        wait_async0();
        __syncthreads();
    }

    #pragma unroll
    for (int t = 0; t < 4; ++t)
        #pragma unroll
        for (int g = 0; g < 8; ++g) {
            const int row = q0 + g + hs * 8;
            const float o = acc[t][g] / l[g];
            attn[((size_t)(b * SS + row)) * DD + h * HD + t * 16 + r] =
                (_Float16)o;
        }
}

// ---------- GEMM2: out = attn @ out_w + out_b (f32 output) ----------
__global__ void __launch_bounds__(128)
out_gemm_kernel(const _Float16* __restrict__ A,   // attn f16 [8192,1024]
                const _Float16* __restrict__ BT,  // out_w^T f16 [1024,1024]
                const float*    __restrict__ bias,
                float* __restrict__ out)
{
    __shared__ __align__(16) _Float16 ldsB[2][64][32];

    const int tid  = threadIdx.x;
    const int lane = tid & 31;
    const int wave = tid >> 5;
    const int n0   = blockIdx.x * 64;
    const int m0   = blockIdx.y * 128 + wave * 32;
    const int r    = lane & 15;
    const int lh   = (lane & 16) ? 8 : 0;
    const int hs   = (lane >> 4) & 1;

    const unsigned ldsb = (unsigned)(uintptr_t)&ldsB[0][0][0];
    const _Float16* arow0 = A + (size_t)(m0 + r) * DD;
    const _Float16* arow1 = A + (size_t)(m0 + 16 + r) * DD;

    async_fill_tile(ldsb, (const char*)(BT + (size_t)n0 * DD), 6, (size_t)DD * 2, tid);
    wait_async0();
    __syncthreads();

    v16h a0c = load_frag(arow0, lh);
    v16h a1c = load_frag(arow1, lh);

    v8f acc0[4] = {}, acc1[4] = {};

    for (int kc = 0; kc < DD; kc += 32) {
        const int  buf  = (kc >> 5) & 1;
        const bool more = (kc + 32) < DD;
        if (more)
            async_fill_tile(ldsb + (buf ^ 1) * 4096,
                            (const char*)(BT + (size_t)n0 * DD + kc + 32),
                            6, (size_t)DD * 2, tid);
        v16h a0n = a0c, a1n = a1c;
        if (more) {
            a0n = load_frag(arow0 + kc + 32, lh);
            a1n = load_frag(arow1 + kc + 32, lh);
        }
        #pragma unroll
        for (int nt = 0; nt < 4; ++nt) {
            v16h bf = load_frag_lds(&ldsB[buf][nt * 16 + r][0], lh);
            acc0[nt] = wmma_f16(a0c, bf, acc0[nt]);
            acc1[nt] = wmma_f16(a1c, bf, acc1[nt]);
        }
        a0c = a0n; a1c = a1n;
        wait_async0();
        __syncthreads();
    }

    #pragma unroll
    for (int nt = 0; nt < 4; ++nt) {
        const int col  = n0 + nt * 16 + r;
        const float bc = bias[col];
        #pragma unroll
        for (int mt = 0; mt < 2; ++mt) {
            const v8f* accp = mt ? acc1 : acc0;
            #pragma unroll
            for (int g = 0; g < 8; ++g) {
                const int row = m0 + mt * 16 + g + hs * 8;
                out[(size_t)row * DD + col] = accp[nt][g] + bc;
            }
        }
    }
}

// ---------- launcher ----------
extern "C" void kernel_launch(void* const* d_in, const int* in_sizes, int n_in,
                              void* d_out, int out_size, void* d_ws, size_t ws_size,
                              hipStream_t stream) {
    const float* src   = (const float*)d_in[0];
    const float* qkv_w = (const float*)d_in[1];
    const float* qkv_b = (const float*)d_in[2];
    const float* out_w = (const float*)d_in[3];
    const float* out_b = (const float*)d_in[4];
    float* out = (float*)d_out;

    char* ws = (char*)d_ws;
    size_t off = 0;
    auto carve = [&](size_t bytes) -> char* {
        char* p = ws + off;
        off = (off + bytes + 255) & ~(size_t)255;
        return p;
    };
    _Float16* srch   = (_Float16*)carve((size_t)M_TOT * DD * 2);        // 16 MB
    _Float16* qkvwT  = (_Float16*)carve((size_t)N_QKV * DD * 2);        //  6 MB
    _Float16* outwT  = (_Float16*)carve((size_t)DD * DD * 2);           //  2 MB
    _Float16* qh     = (_Float16*)carve((size_t)BB * HH * SS * HD * 2); // 16 MB
    _Float16* kh     = (_Float16*)carve((size_t)BB * HH * SS * HD * 2); // 16 MB
    _Float16* vth    = (_Float16*)carve((size_t)BB * HH * HD * SS * 2); // 16 MB
    _Float16* attnh  = (_Float16*)carve((size_t)M_TOT * DD * 2);        // 16 MB

    {
        int n = M_TOT * DD;
        cvt_f32_to_f16<<<(n + 255) / 256, 256, 0, stream>>>(src, srch, n);
    }
    {
        int n = N_QKV * DD;
        transpose_cvt<<<(n + 255) / 256, 256, 0, stream>>>(qkv_w, qkvwT, DD, N_QKV);
    }
    {
        int n = DD * DD;
        transpose_cvt<<<(n + 255) / 256, 256, 0, stream>>>(out_w, outwT, DD, DD);
    }

    qkv_gemm_kernel<<<dim3(N_QKV / 64, M_TOT / 128), 128, 0, stream>>>(
        srch, qkvwT, qkv_b, qh, kh, vth);

    attn_kernel<<<dim3(SS / 64, BB * HH), 128, 0, stream>>>(qh, kh, vth, attnh);

    out_gemm_kernel<<<dim3(DD / 64, M_TOT / 128), 128, 0, stream>>>(
        attnh, outwT, out_b, out);
}
